// MultiHeadLatentAttention_52106543235529
// MI455X (gfx1250) — matrix operand order repaired and checked by reading, not compiled, not run on hardware
//
#include <hip/hip_runtime.h>

typedef __attribute__((ext_vector_type(16))) __bf16 v16bf;
typedef __attribute__((ext_vector_type(8)))  __bf16 v8bf;
typedef __attribute__((ext_vector_type(8)))  float  v8f;

#define NH   9
#define HD   64
#define HID  576
#define LAT  144
#define BB   4
#define SS   2048
#define ROWS (BB * SS)   // 8192

// ---------- bf16 helpers (round-to-nearest-even) ----------
__device__ __forceinline__ __bf16 f2bf(float f) {
  union { float f; unsigned u; } in{f};
  unsigned u = in.u;
  unsigned r = (u + 0x7FFFu + ((u >> 16) & 1u)) >> 16;
  union { unsigned short s; __bf16 b; } out;
  out.s = (unsigned short)r;
  return out.b;
}
__device__ __forceinline__ float bf2f(__bf16 b) {
  union { __bf16 b; unsigned short s; } in{b};
  union { unsigned u; float f; } out{(unsigned)in.s << 16};
  return out.f;
}

// ---------- one-time f32 -> bf16 conversion ----------
__global__ __launch_bounds__(256)
void convert_bf16(const float* __restrict__ in, __bf16* __restrict__ out, int n) {
  int idx = blockIdx.x * blockDim.x + threadIdx.x;
  if (idx < n) out[idx] = f2bf(in[idx]);
}

// ---------- WMMA GEMM: C[M,N] = A[M,K] * W[N,K]^T, all-bf16 operands ----------
// one wave computes a 16x64 C strip (4 n-tiles): A fragment reused 4x per k-step,
// inner loop is pure b128 loads + v_wmma (no conversion VALU).
template<bool OUT_BF16>
__global__ __launch_bounds__(256)
void gemm_bf16(const __bf16* __restrict__ A, const __bf16* __restrict__ W,
               void* __restrict__ Craw, int M, int N, int K) {
  const int lane = threadIdx.x & 31;
  const int wave = threadIdx.x >> 5;
  const int half = lane >> 4;
  const int lm   = lane & 15;
  const int ntiles  = N >> 4;
  const int ngroups = (ntiles + 3) >> 2;
  const int total   = (M >> 4) * ngroups;
  int tile = blockIdx.x * 8 + wave;
  if (tile >= total) return;
  const int mt = tile / ngroups;
  const int ng = tile - mt * ngroups;
  const int nt0 = ng * 4;

  v8f acc[4] = {};
  const __bf16* arow = A + (size_t)(mt * 16 + lm) * (size_t)K;  // A-layout lane row

  for (int k0 = 0; k0 < K; k0 += 32) {
    v8bf z8 = {};
    union { v16bf v; v8bf h[2]; } af;
    const int ka0 = k0 + half * 16;       // A element j -> k = ka0 + j
    af.h[0] = (ka0 < K) ? *(const v8bf*)(arow + ka0) : z8;
    af.h[1] = (ka0 < K) ? *(const v8bf*)(arow + ka0 + 8) : z8;
    const int kw = k0 + half * 16;        // B element j -> k = kw + j
#pragma unroll
    for (int t = 0; t < 4; ++t) {
      const int nt = nt0 + t;
      if (nt >= ntiles) break;
      const __bf16* wr = W + (size_t)(nt * 16 + lm) * (size_t)K;
      union { v16bf v; v8bf h[2]; } bf;
      if (kw < K) {
        bf.h[0] = *(const v8bf*)(wr + kw);
        bf.h[1] = *(const v8bf*)(wr + kw + 8);
      } else { bf.h[0] = z8; bf.h[1] = z8; }
      acc[t] = __builtin_amdgcn_wmma_f32_16x16x32_bf16(false, af.v, false, bf.v,
                                                       (short)0, acc[t], false, false);
    }
  }

#pragma unroll
  for (int t = 0; t < 4; ++t) {
    const int nt = nt0 + t;
    if (nt >= ntiles) break;
    const int col = nt * 16 + lm;
#pragma unroll
    for (int r = 0; r < 8; ++r) {
      const size_t row = (size_t)(mt * 16 + half * 8 + r);
      if (OUT_BF16) ((__bf16*)Craw)[row * (size_t)N + col] = f2bf(acc[t][r]);
      else          ((float *)Craw)[row * (size_t)N + col] = acc[t][r];
    }
  }
}

// ---------- RoPE (in place, bf16 q/k buffers of shape [ROWS, HID]) ----------
__global__ __launch_bounds__(256)
void rope_kernel(__bf16* __restrict__ buf) {
  int idx = blockIdx.x * blockDim.x + threadIdx.x;
  const int total = ROWS * NH * (HD / 2);
  if (idx >= total) return;
  int d   = idx & 31;
  int t   = idx >> 5;
  int h   = t % NH;
  int row = t / NH;
  int s   = row & (SS - 1);
  float inv = __powf(10000.0f, -(float)(2 * d) * (1.0f / 64.0f));
  float ang = (float)s * inv;
  float sn, cs;
  __sincosf(ang, &sn, &cs);
  __bf16* p = buf + (size_t)row * HID + h * HD + d;
  float x1 = bf2f(p[0]);
  float x2 = bf2f(p[32]);
  p[0]  = f2bf(x1 * cs - x2 * sn);
  p[32] = f2bf(x2 * cs + x1 * sn);
}

// ---------- V transpose: [b*S, h*64+d] -> [b,h,d,s] ----------
__global__ __launch_bounds__(256)
void transpose_v(const __bf16* __restrict__ V, __bf16* __restrict__ Vt) {
  int idx = blockIdx.x * blockDim.x + threadIdx.x;
  if (idx >= ROWS * HID) return;
  int col = idx % HID;
  int row = idx / HID;
  int h = col / HD, d = col % HD;
  int b = row / SS, s = row % SS;
  Vt[((size_t)(b * NH + h) * HD + d) * SS + s] = V[(size_t)row * HID + col];
}

// ---------- causal flash attention, one wave per 16-row q tile ----------
__global__ __launch_bounds__(256)
void flash_kernel(const __bf16* __restrict__ Q, const __bf16* __restrict__ Kbuf,
                  const __bf16* __restrict__ Vt, __bf16* __restrict__ Ctx) {
  __shared__ __bf16 plds[8][16][32];   // per-wave P staging (C-layout -> A-layout)
  const int lane = threadIdx.x & 31;
  const int wave = threadIdx.x >> 5;
  const int half = lane >> 4;
  const int lm   = lane & 15;
  const int QT   = SS / 16;
  int wid = blockIdx.x * 8 + wave;
  if (wid >= BB * NH * QT) return;
  const int qt = wid % QT;
  const int bh = wid / QT;
  const int h  = bh % NH;
  const int b  = bh / NH;

  // Q fragments (16x64 bf16, two K=32 A-operands), A-layout loads
  const __bf16* qp = Q + (size_t)(b * SS + qt * 16 + lm) * HID + h * HD;
  union { v16bf v; v8bf hh[2]; } a0, a1;
  a0.hh[0] = *(const v8bf*)(qp + half * 16);
  a0.hh[1] = *(const v8bf*)(qp + half * 16 + 8);
  a1.hh[0] = *(const v8bf*)(qp + 32 + half * 16);
  a1.hh[1] = *(const v8bf*)(qp + 32 + half * 16 + 8);

  float m_i[8], l_i[8];
  v8f acc0 = {}, acc1 = {}, acc2 = {}, acc3 = {};
#pragma unroll
  for (int r = 0; r < 8; ++r) { m_i[r] = -INFINITY; l_i[r] = 0.0f; }

  const float scale = 0.125f;           // 1/sqrt(64)
  const int qmax = qt * 16 + 15;

  for (int kb = 0; kb <= qmax; kb += 32) {
    v8f s0 = {}, s1 = {};
    {   // score tile 0: keys kb..kb+15; B = K^T (head-dim on K axis)
      const __bf16* kp = Kbuf + (size_t)(b * SS + kb + lm) * HID + h * HD;
      union { v16bf v; v8bf hh[2]; } b0, b1;
      b0.hh[0] = *(const v8bf*)(kp + half * 16);
      b0.hh[1] = *(const v8bf*)(kp + half * 16 + 8);
      b1.hh[0] = *(const v8bf*)(kp + 32 + half * 16);
      b1.hh[1] = *(const v8bf*)(kp + 32 + half * 16 + 8);
      s0 = __builtin_amdgcn_wmma_f32_16x16x32_bf16(false, a0.v, false, b0.v, (short)0, s0, false, false);
      s0 = __builtin_amdgcn_wmma_f32_16x16x32_bf16(false, a1.v, false, b1.v, (short)0, s0, false, false);
    }
    {   // score tile 1: keys kb+16..kb+31
      const __bf16* kp = Kbuf + (size_t)(b * SS + kb + 16 + lm) * HID + h * HD;
      union { v16bf v; v8bf hh[2]; } b0, b1;
      b0.hh[0] = *(const v8bf*)(kp + half * 16);
      b0.hh[1] = *(const v8bf*)(kp + half * 16 + 8);
      b1.hh[0] = *(const v8bf*)(kp + 32 + half * 16);
      b1.hh[1] = *(const v8bf*)(kp + 32 + half * 16 + 8);
      s1 = __builtin_amdgcn_wmma_f32_16x16x32_bf16(false, a0.v, false, b0.v, (short)0, s1, false, false);
      s1 = __builtin_amdgcn_wmma_f32_16x16x32_bf16(false, a1.v, false, b1.v, (short)0, s1, false, false);
    }

    // online softmax; C-layout: lane holds col n=lm, rows m = r + half*8.
    // xor reductions over {1,2,4,8} stay inside each 16-lane half-group.
#pragma unroll
    for (int r = 0; r < 8; ++r) {
      const int qg = qt * 16 + half * 8 + r;
      float v0 = (kb + lm      > qg) ? -INFINITY : s0[r] * scale;
      float v1 = (kb + 16 + lm > qg) ? -INFINITY : s1[r] * scale;
      float mx = fmaxf(v0, v1);
#pragma unroll
      for (int o = 8; o >= 1; o >>= 1) mx = fmaxf(mx, __shfl_xor(mx, o, 32));
      const float mnew = fmaxf(m_i[r], mx);
      const float p0 = __expf(v0 - mnew);
      const float p1 = __expf(v1 - mnew);
      float rs = p0 + p1;
#pragma unroll
      for (int o = 8; o >= 1; o >>= 1) rs += __shfl_xor(rs, o, 32);
      const float alpha = __expf(m_i[r] - mnew);
      m_i[r] = mnew;
      l_i[r] = l_i[r] * alpha + rs;
      acc0[r] *= alpha; acc1[r] *= alpha; acc2[r] *= alpha; acc3[r] *= alpha;
      plds[wave][half * 8 + r][lm]      = f2bf(p0);
      plds[wave][half * 8 + r][16 + lm] = f2bf(p1);
    }
    // per-wave LDS RAW fence (cross-lane re-layout; no block barrier: waves diverge)
    asm volatile("s_wait_dscnt 0x0" ::: "memory");

    // gather P as A-operand (16x32)
    union { v16bf v; v8bf hh[2]; } pa;
    pa.hh[0] = *(const v8bf*)(&plds[wave][lm][half * 16]);
    pa.hh[1] = *(const v8bf*)(&plds[wave][lm][half * 16 + 8]);

    // P(16x32) x V(32x64): V^T layout makes the K-dim (keys) contiguous
    const __bf16* vbase = Vt + (size_t)(b * NH + h) * HD * SS + kb + half * 16;
#pragma unroll
    for (int n4 = 0; n4 < 4; ++n4) {
      const __bf16* vp = vbase + (size_t)(n4 * 16 + lm) * SS;
      union { v16bf v; v8bf hh[2]; } vbf;
      vbf.hh[0] = *(const v8bf*)(vp);
      vbf.hh[1] = *(const v8bf*)(vp + 8);
      v8f* accp = (n4 == 0) ? &acc0 : (n4 == 1) ? &acc1 : (n4 == 2) ? &acc2 : &acc3;
      *accp = __builtin_amdgcn_wmma_f32_16x16x32_bf16(false, pa.v, false, vbf.v,
                                                      (short)0, *accp, false, false);
    }
    asm volatile("s_wait_dscnt 0x0" ::: "memory");   // WAR before next block's P writes
  }

  __bf16* op = Ctx + (size_t)(b * SS + qt * 16 + half * 8) * HID + h * HD + lm;
#pragma unroll
  for (int r = 0; r < 8; ++r) {
    const float inv_l = 1.0f / l_i[r];
    op[(size_t)r * HID + 0]  = f2bf(acc0[r] * inv_l);
    op[(size_t)r * HID + 16] = f2bf(acc1[r] * inv_l);
    op[(size_t)r * HID + 32] = f2bf(acc2[r] * inv_l);
    op[(size_t)r * HID + 48] = f2bf(acc3[r] * inv_l);
  }
}

// ---------- launch ----------
extern "C" void kernel_launch(void* const* d_in, const int* in_sizes, int n_in,
                              void* d_out, int out_size, void* d_ws, size_t ws_size,
                              hipStream_t stream) {
  const float* x    = (const float*)d_in[0];
  const float* Wqd  = (const float*)d_in[1];
  const float* Wkvd = (const float*)d_in[2];
  const float* Wqu  = (const float*)d_in[3];
  const float* Wku  = (const float*)d_in[4];
  const float* Wvu  = (const float*)d_in[5];
  const float* Wo   = (const float*)d_in[6];

  char* ws = (char*)d_ws;
  __bf16* xb     = (__bf16*)(ws + 0);         // 8192*576 bf16 = 9.44 MB
  __bf16* q_lat  = (__bf16*)(ws + 9437184);   // 8192*144 bf16 = 2.36 MB
  __bf16* kv_lat = (__bf16*)(ws + 11796480);
  __bf16* qb     = (__bf16*)(ws + 14155776);  // 8192*576 bf16 each
  __bf16* kbuf   = (__bf16*)(ws + 23592960);
  __bf16* vbuf   = (__bf16*)(ws + 33030144);
  __bf16* vt     = (__bf16*)(ws + 42467328);
  __bf16* ctx    = (__bf16*)(ws + 51904512);
  __bf16* bWqd   = (__bf16*)(ws + 61341696);  // 144*576 = 82944 elems each
  __bf16* bWkvd  = (__bf16*)(ws + 61507584);
  __bf16* bWqu   = (__bf16*)(ws + 61673472);
  __bf16* bWku   = (__bf16*)(ws + 61839360);
  __bf16* bWvu   = (__bf16*)(ws + 62005248);
  __bf16* bWo    = (__bf16*)(ws + 62171136);  // 576*576 = 331776 elems

  dim3 blk(256);
  // one-time bf16 conversion of activations + weights
  convert_bf16<<<(ROWS * HID + 255) / 256, blk, 0, stream>>>(x,    xb,    ROWS * HID);
  convert_bf16<<<(LAT * HID + 255) / 256, blk, 0, stream>>>(Wqd,  bWqd,  LAT * HID);
  convert_bf16<<<(LAT * HID + 255) / 256, blk, 0, stream>>>(Wkvd, bWkvd, LAT * HID);
  convert_bf16<<<(HID * LAT + 255) / 256, blk, 0, stream>>>(Wqu,  bWqu,  HID * LAT);
  convert_bf16<<<(HID * LAT + 255) / 256, blk, 0, stream>>>(Wku,  bWku,  HID * LAT);
  convert_bf16<<<(HID * LAT + 255) / 256, blk, 0, stream>>>(Wvu,  bWvu,  HID * LAT);
  convert_bf16<<<(HID * HID + 255) / 256, blk, 0, stream>>>(Wo,   bWo,   HID * HID);

  // down projections: M=8192, N=144 (3 n-groups), K=576 -> 1536 waves -> 192 blocks
  gemm_bf16<true><<<192, blk, 0, stream>>>(xb, bWqd,  q_lat,  ROWS, LAT, HID);
  gemm_bf16<true><<<192, blk, 0, stream>>>(xb, bWkvd, kv_lat, ROWS, LAT, HID);
  // up projections: M=8192, N=576 (9 n-groups), K=144 -> 4608 waves -> 576 blocks
  gemm_bf16<true><<<576, blk, 0, stream>>>(q_lat,  bWqu, qb,   ROWS, HID, LAT);
  gemm_bf16<true><<<576, blk, 0, stream>>>(kv_lat, bWku, kbuf, ROWS, HID, LAT);
  gemm_bf16<true><<<576, blk, 0, stream>>>(kv_lat, bWvu, vbuf, ROWS, HID, LAT);
  // RoPE on q and k
  rope_kernel<<<9216, blk, 0, stream>>>(qb);
  rope_kernel<<<9216, blk, 0, stream>>>(kbuf);
  // V transpose for contiguous P*V operands
  transpose_v<<<18432, blk, 0, stream>>>(vbuf, vt);
  // flash attention: 4*9*128 = 4608 waves -> 576 blocks
  flash_kernel<<<576, blk, 0, stream>>>(qb, kbuf, vt, ctx);
  // output projection: M=8192, N=576, K=576 -> 576 blocks
  gemm_bf16<false><<<576, blk, 0, stream>>>(ctx, bWo, d_out, ROWS, HID, HID);
}
